// XENetGCN_47218870452990
// MI455X (gfx1250) — matrix operand order repaired and checked by reading, not compiled, not run on hardware
//
#include <hip/hip_runtime.h>

typedef __bf16 bf16;
typedef __attribute__((ext_vector_type(8)))  __bf16 v8bf;
typedef __attribute__((ext_vector_type(16))) __bf16 v16bf;
typedef __attribute__((ext_vector_type(8)))  float  v8f;

#define NN     256
#define DIM    768
#define E_IN   32
#define NE     65280      // 256*255 edges
#define KCAT   2304       // 3*DIM
#define MT     64         // edge rows per fused-conv workgroup
#define RP     776        // LDS stack row pitch (bf16 elems): 768 + 8 pad
#define BROW   128        // staged-B k-width per superstep (elems)

// ---- WMMA fragment helpers (layouts per CDNA5 ISA 7.12.2) -----------------
static __device__ __forceinline__ v16bf ld_a(const bf16* row, int half) {
  v8bf lo = *(const v8bf*)(row + half * 8);
  v8bf hi = *(const v8bf*)(row + half * 8 + 16);
  return __builtin_shufflevector(lo, hi, 0,1,2,3,4,5,6,7,8,9,10,11,12,13,14,15);
}
static __device__ __forceinline__ v8f wmma_bf(v16bf a, v16bf b, v8f c) {
  return __builtin_amdgcn_wmma_f32_16x16x32_bf16(false, a, false, b, (short)0, c,
                                                 false, false);
}

// ---- fast activations: branch-free exp/rcp (v_exp_f32 + v_rcp_f32) --------
static __device__ __forceinline__ float fast_sigmoid(float x) {
  return __builtin_amdgcn_rcpf(1.f + __expf(-x));
}
static __device__ __forceinline__ float fast_tanh(float x) {
  x = fminf(fmaxf(x, -15.f), 15.f);
  float e = __expf(2.f * x);
  return (e - 1.f) * __builtin_amdgcn_rcpf(e + 1.f);
}

// ---- CDNA5 async copy: global -> LDS, 16B per lane, tracked by ASYNCcnt ---
static __device__ __forceinline__ void async_ld16(const bf16* g, void* l) {
  unsigned loff = (unsigned)(uintptr_t)l;       // generic LDS addr low 32 bits
  unsigned long long ga = (unsigned long long)(uintptr_t)g;
  asm volatile("global_load_async_to_lds_b128 %0, %1, off"
               :: "v"(loff), "v"(ga) : "memory");
}
template <int N>
static __device__ __forceinline__ void wait_asynccnt() {
  asm volatile("s_wait_asynccnt %0" :: "n"(N) : "memory");
}

// stage a 64-col x 128-k bf16 weight tile (16KB) into LDS: 8 x 16B per thread
static __device__ __forceinline__ void stage_b(const bf16* __restrict__ Wt,
                                               int K, int n0, int k0,
                                               bf16* sbuf, int tid) {
#pragma unroll
  for (int it = 0; it < 8; ++it) {
    int q = tid + it * 128;            // 0..1023 -> 1024 x 16B = 16KB
    int rown = q >> 4;                 // 0..63
    int seg  = q & 15;                 // 16B segment in 256B row
    async_ld16(Wt + (size_t)(n0 + rown) * K + k0 + seg * 8,
               sbuf + rown * BROW + seg * 8);
  }
}

// ---- small utility kernels -------------------------------------------------
__global__ void k_cvt_x(const float* __restrict__ x, bf16* __restrict__ xbf,
                        float* __restrict__ xcur) {
  int i = blockIdx.x * 256 + threadIdx.x;
  if (i < NN * DIM) { float v = x[i]; xbf[i] = (bf16)v; xcur[i] = v; }
}

__global__ void k_wt(const float* __restrict__ W, bf16* __restrict__ Wt,
                     int din, int dout) {
  int i = blockIdx.x * 256 + threadIdx.x;
  if (i < din * dout) {
    int r = i / dout, c = i - r * dout;
    Wt[(size_t)c * din + r] = (bf16)W[i];
  }
}

__global__ void k_build_xcat(const float* __restrict__ xcur,
                             const float* __restrict__ oi,
                             const float* __restrict__ oj,
                             bf16* __restrict__ xcat) {
  int i = blockIdx.x * 256 + threadIdx.x;
  if (i < NN * KCAT) {
    int m = i / KCAT, k = i - m * KCAT;
    float v = (k < DIM)     ? xcur[m * DIM + k]
            : (k < 2 * DIM) ? oi[m * DIM + k - DIM]
                            : oj[m * DIM + k - 2 * DIM];
    xcat[i] = (bf16)v;
  }
}

// ---- pre-edge: ebf = tanh(e @ Wpre + b), one 16x16 tile per wave, K=32 ----
__global__ __launch_bounds__(128)
void k_pre_edge(const float* __restrict__ e, const bf16* __restrict__ WtP,
                const float* __restrict__ bpre, bf16* __restrict__ ebf) {
  int lane = threadIdx.x & 31, wave = threadIdx.x >> 5;
  int half = lane >> 4, l16 = lane & 15;
  int tile = blockIdx.x * 4 + wave;
  int mt = tile / 48, nt = tile - mt * 48;
  int m0 = mt * 16;
  const float* er = e + (size_t)(m0 + l16) * E_IN;
  v16bf a;
#pragma unroll
  for (int u = 0; u < 8; ++u) {
    a[u]     = (bf16)er[half * 8 + u];
    a[u + 8] = (bf16)er[half * 8 + 16 + u];
  }
  int n = nt * 16 + l16;
  v16bf b = ld_a(WtP + (size_t)n * E_IN, half);
  v8f c = {};
  c = wmma_bf(a, b, c);
  float bia = bpre[n];
  int mloc = half * 8;
#pragma unroll
  for (int r = 0; r < 8; ++r)
    ebf[(size_t)(m0 + mloc + r) * DIM + n] = (bf16)fast_tanh(c[r] + bia);
}

// ---- fused XENet conv: stack GEMM + leakyReLU + att + segsum + edge GEMM --
__global__ __launch_bounds__(128)
void k_conv_fused(const bf16* __restrict__ xbf, bf16* __restrict__ ebf,
                  const int* __restrict__ src, const int* __restrict__ tgt,
                  const bf16* __restrict__ WtS, const float* __restrict__ bS,
                  const float* __restrict__ alpha,
                  const float* __restrict__ wai, const float* __restrict__ bai,
                  const float* __restrict__ waj, const float* __restrict__ baj,
                  const bf16* __restrict__ WtE, const float* __restrict__ bE,
                  float* __restrict__ out_i, float* __restrict__ out_j) {
  __shared__ bf16  sstack[MT * RP];          // 99.3 KB stack-activation tile
  __shared__ bf16  sB[2][64 * BROW];         // 2 x 16 KB staged weight tiles
  __shared__ float satti[MT], sattj[MT];
  __shared__ int   ssrc[MT], stgt[MT];

  int tid  = threadIdx.x;
  int lane = tid & 31, wave = tid >> 5;
  int half = lane >> 4, l16 = lane & 15;
  int m0   = blockIdx.x * MT;
  int rowl = wave * 16 + l16;                // local row this lane feeds as A
  int row  = m0 + rowl;
  const bf16* xs = xbf + (size_t)src[row] * DIM;
  const bf16* xt = xbf + (size_t)tgt[row] * DIM;
  const bf16* ee = ebf + (size_t)row * DIM;
  int mloc = wave * 16 + half * 8;

  // ---- stack GEMM: 12 chunks of 64 cols; B double-buffered via async copy --
  for (int nc = 0; nc < 12; ++nc) {
    int n0 = nc * 64;
    v8f acc[4]; acc[0] = {}; acc[1] = {}; acc[2] = {}; acc[3] = {};
    stage_b(WtS, KCAT, n0, 0, sB[0], tid);
    for (int ss = 0; ss < KCAT / BROW; ++ss) {          // 18 supersteps
      bf16* cur = sB[ss & 1];
      if (ss + 1 < KCAT / BROW) {
        stage_b(WtS, KCAT, n0, (ss + 1) * BROW, sB[(ss + 1) & 1], tid);
        wait_asynccnt<8>();                 // oldest batch (cur) retired
      } else {
        wait_asynccnt<0>();
      }
      __syncthreads();                      // cur visible to all waves
#pragma unroll
      for (int k2 = 0; k2 < 4; ++k2) {
        int k0 = ss * BROW + k2 * 32;
        const bf16* ap = (k0 < DIM)     ? xs + k0
                       : (k0 < 2 * DIM) ? xt + (k0 - DIM)
                                        : ee + (k0 - 2 * DIM);
        v16bf a = ld_a(ap, half);
#pragma unroll
        for (int j = 0; j < 4; ++j) {
          v16bf b = ld_a(cur + (j * 16 + l16) * BROW + k2 * 32, half);
          acc[j] = wmma_bf(a, b, acc[j]);
        }
      }
      __syncthreads();                      // done reading cur before reuse
    }
#pragma unroll
    for (int j = 0; j < 4; ++j) {
      int n = n0 + j * 16 + l16;
      float bia = bS[n], al = alpha[n];
#pragma unroll
      for (int r = 0; r < 8; ++r) {
        float v = acc[j][r] + bia;
        v = (v >= 0.f) ? v : al * v;        // leaky-ReLU(alpha[n])
        sstack[(mloc + r) * RP + n] = (bf16)v;
      }
    }
  }
  __syncthreads();

  // ---- attention gates: one row per thread, vectorized LDS reads ----------
  if (tid < MT) {
    int gr = m0 + tid;
    ssrc[tid] = src[gr]; stgt[tid] = tgt[gr];
    const v8bf* sr = (const v8bf*)(sstack + tid * RP);
    float di = 0.f, dj = 0.f;
    for (int k = 0; k < DIM / 8; ++k) {
      v8bf v8 = sr[k];
#pragma unroll
      for (int u = 0; u < 8; ++u) {
        float v = (float)v8[u];
        di += v * wai[k * 8 + u]; dj += v * waj[k * 8 + u];
      }
    }
    satti[tid] = fast_sigmoid(di + bai[0]);
    sattj[tid] = fast_sigmoid(dj + baj[0]);
  }
  __syncthreads();

  // ---- segment sums: out_i reduced per-block (<=2 distinct src),
  //      out_j per-element f32 atomics --------------------------------------
  int s0 = ssrc[0], s1 = ssrc[MT - 1];
  for (int cn = tid; cn < DIM; cn += 128) {
    float a0 = 0.f, a1 = 0.f;
    for (int rm = 0; rm < MT; ++rm) {
      float v = satti[rm] * (float)sstack[rm * RP + cn];
      if (ssrc[rm] == s0) a0 += v; else a1 += v;
    }
    atomicAdd(&out_i[(size_t)s0 * DIM + cn], a0);
    if (s1 != s0) atomicAdd(&out_i[(size_t)s1 * DIM + cn], a1);
  }
  for (int idx = tid; idx < MT * DIM; idx += 128) {
    int rm = idx / DIM, cn = idx - rm * DIM;
    atomicAdd(&out_j[(size_t)stgt[rm] * DIM + cn],
              sattj[rm] * (float)sstack[rm * RP + cn]);
  }
  __syncthreads();

  // ---- edge GEMM out of LDS, in-place into ebf (own rows only) ------------
  for (int nc = 0; nc < 12; ++nc) {
    int n0 = nc * 64;
    v8f acc[4]; acc[0] = {}; acc[1] = {}; acc[2] = {}; acc[3] = {};
    stage_b(WtE, DIM, n0, 0, sB[0], tid);
    for (int ss = 0; ss < DIM / BROW; ++ss) {           // 6 supersteps
      bf16* cur = sB[ss & 1];
      if (ss + 1 < DIM / BROW) {
        stage_b(WtE, DIM, n0, (ss + 1) * BROW, sB[(ss + 1) & 1], tid);
        wait_asynccnt<8>();
      } else {
        wait_asynccnt<0>();
      }
      __syncthreads();
#pragma unroll
      for (int k2 = 0; k2 < 4; ++k2) {
        int k0 = ss * BROW + k2 * 32;
        v16bf a = ld_a(sstack + rowl * RP + k0, half);
#pragma unroll
        for (int j = 0; j < 4; ++j) {
          v16bf b = ld_a(cur + (j * 16 + l16) * BROW + k2 * 32, half);
          acc[j] = wmma_bf(a, b, acc[j]);
        }
      }
      __syncthreads();
    }
#pragma unroll
    for (int j = 0; j < 4; ++j) {
      int n = n0 + j * 16 + l16;
      float bia = bE[n];
#pragma unroll
      for (int r = 0; r < 8; ++r)
        ebf[(size_t)(m0 + mloc + r) * DIM + n] = (bf16)fast_tanh(acc[j][r] + bia);
    }
  }
}

// ---- node GEMM: x_new = relu(xcat @ Wnode + b), one 16x16 tile per wave ---
__global__ __launch_bounds__(128)
void k_node_gemm(const bf16* __restrict__ xcat, const bf16* __restrict__ WtN,
                 const float* __restrict__ bn, float* __restrict__ xcur,
                 bf16* __restrict__ xbf) {
  int lane = threadIdx.x & 31, wave = threadIdx.x >> 5;
  int half = lane >> 4, l16 = lane & 15;
  int tile = blockIdx.x * 4 + wave;
  int mt = tile / 48, nt = tile - mt * 48;
  int m0 = mt * 16, n = nt * 16 + l16;
  const bf16* arow = xcat + (size_t)(m0 + l16) * KCAT;
  const bf16* bcol = WtN + (size_t)n * KCAT;
  v8f c = {};
  for (int ks = 0; ks < KCAT / 32; ++ks)
    c = wmma_bf(ld_a(arow + ks * 32, half),
                ld_a(bcol + ks * 32, half), c);
  float bia = bn[n];
  int mloc = half * 8;
#pragma unroll
  for (int r = 0; r < 8; ++r) {
    float v = c[r] + bia; v = v > 0.f ? v : 0.f;
    size_t o = (size_t)(m0 + mloc + r) * DIM + n;
    xcur[o] = v; xbf[o] = (bf16)v;
  }
}

// ---- final pooling + logits + softmax -------------------------------------
__global__ __launch_bounds__(256)
void k_final(const float* __restrict__ xcur, const bf16* __restrict__ ebf,
             const float* __restrict__ Wo, const float* __restrict__ bo,
             float* __restrict__ out) {
  __shared__ float pooled[4 * DIM];
  __shared__ float logits[16];
  int tid = threadIdx.x;
  for (int d = tid; d < DIM; d += 256) {
    float p1 = 0.f, p2 = 0.f, p3 = 0.f;
    for (int m = 1; m < NN; ++m) p1 += xcur[m * DIM + d];
    for (int r = 0; r < NN - 1; ++r) p2 += (float)ebf[(size_t)r * DIM + d];
    for (int k = 0; k < NN - 1; ++k)
      p3 += (float)ebf[(size_t)(k * (NN - 1)) * DIM + d];
    pooled[d]           = 255.f * xcur[d];
    pooled[DIM + d]     = p1;
    pooled[2 * DIM + d] = p2;
    pooled[3 * DIM + d] = p3;
  }
  __syncthreads();
  if (tid < 16) {
    float acc = bo[tid];
    for (int k = 0; k < 4 * DIM; ++k) acc += pooled[k] * Wo[k * 16 + tid];
    logits[tid] = acc;
  }
  __syncthreads();
  if (tid == 0) {
    float mx = logits[0];
    for (int i = 1; i < 16; ++i) mx = fmaxf(mx, logits[i]);
    float s = 0.f, ex[16];
    for (int i = 0; i < 16; ++i) { ex[i] = __expf(logits[i] - mx); s += ex[i]; }
    for (int i = 0; i < 16; ++i) out[i] = ex[i] / s;
  }
}

// ---------------------------------------------------------------------------
extern "C" void kernel_launch(void* const* d_in, const int* in_sizes, int n_in,
                              void* d_out, int out_size, void* d_ws,
                              size_t ws_size, hipStream_t stream) {
  (void)in_sizes; (void)n_in; (void)out_size; (void)ws_size;
  const float* x    = (const float*)d_in[0];
  const float* e    = (const float*)d_in[1];
  const int*   src  = (const int*)d_in[2];
  const int*   tgt  = (const int*)d_in[3];
  const float* preW = (const float*)d_in[4];
  const float* preB = (const float*)d_in[5];
  const float* outW = (const float*)d_in[28];
  const float* outB = (const float*)d_in[29];

  char* p = (char*)d_ws;
  auto carve = [&](size_t bytes) -> char* {
    char* r = p; p += (bytes + 255) & ~(size_t)255; return r;
  };
  bf16*  ebf  = (bf16*)carve((size_t)NE * DIM * 2);   // edge acts (in-place)
  bf16*  WtS  = (bf16*)carve((size_t)DIM * KCAT * 2);
  bf16*  WtN  = (bf16*)carve((size_t)DIM * KCAT * 2);
  bf16*  WtE  = (bf16*)carve((size_t)DIM * DIM * 2);
  bf16*  WtP  = (bf16*)carve((size_t)DIM * E_IN * 2);
  bf16*  xbf  = (bf16*)carve((size_t)NN * DIM * 2);
  float* xcur = (float*)carve((size_t)NN * DIM * 4);
  float* oi   = (float*)carve((size_t)NN * DIM * 4);
  float* oj   = (float*)carve((size_t)NN * DIM * 4);
  bf16*  xcat = (bf16*)carve((size_t)NN * KCAT * 2);

  k_cvt_x<<<(NN * DIM + 255) / 256, 256, 0, stream>>>(x, xbf, xcur);
  k_wt<<<(E_IN * DIM + 255) / 256, 256, 0, stream>>>(preW, WtP, E_IN, DIM);
  k_pre_edge<<<(NE / 16) * 48 / 4, 128, 0, stream>>>(e, WtP, preB, ebf);

  for (int c = 0; c < 2; ++c) {
    int b0 = 6 + c * 11;
    const float* sW  = (const float*)d_in[b0 + 0];
    const float* sB  = (const float*)d_in[b0 + 1];
    const float* al  = (const float*)d_in[b0 + 2];
    const float* aiW = (const float*)d_in[b0 + 3];
    const float* aiB = (const float*)d_in[b0 + 4];
    const float* ajW = (const float*)d_in[b0 + 5];
    const float* ajB = (const float*)d_in[b0 + 6];
    const float* nW  = (const float*)d_in[b0 + 7];
    const float* nB  = (const float*)d_in[b0 + 8];
    const float* eW  = (const float*)d_in[b0 + 9];
    const float* eB  = (const float*)d_in[b0 + 10];

    k_wt<<<(KCAT * DIM + 255) / 256, 256, 0, stream>>>(sW, WtS, KCAT, DIM);
    k_wt<<<(DIM * DIM + 255) / 256, 256, 0, stream>>>(eW, WtE, DIM, DIM);
    k_wt<<<(KCAT * DIM + 255) / 256, 256, 0, stream>>>(nW, WtN, KCAT, DIM);
    hipMemsetAsync(oi, 0, (size_t)NN * DIM * 4, stream);
    hipMemsetAsync(oj, 0, (size_t)NN * DIM * 4, stream);

    k_conv_fused<<<NE / MT, 128, 0, stream>>>(xbf, ebf, src, tgt, WtS, sB, al,
                                              aiW, aiB, ajW, ajB, WtE, eB, oi, oj);
    k_build_xcat<<<(NN * KCAT + 255) / 256, 256, 0, stream>>>(xcur, oi, oj, xcat);
    k_node_gemm<<<16 * 48 / 4, 128, 0, stream>>>(xcat, WtN, nB, xcur, xbf);
  }

  k_final<<<1, 256, 0, stream>>>(xcur, ebf, outW, outB, (float*)d_out);
}
